// INNv2CUDA_44976897523879
// MI455X (gfx1250) — matrix operand order repaired and checked by reading, not compiled, not run on hardware
//
#include <hip/hip_runtime.h>
#include <hip/hip_bf16.h>
#include <math.h>

typedef __attribute__((ext_vector_type(16))) _Float16 v16h;
typedef __attribute__((ext_vector_type(8)))  _Float16 v8h;
typedef __attribute__((ext_vector_type(8)))  float    v8f;
typedef __attribute__((ext_vector_type(4)))  float    v4f;

#define L_SEQ 512
#define NN_   8
#define D_    256
#define DI_   512
#define DS_   16
#define DC_   4
#define DTR_  16
#define H_    4
#define NL_   4
#define ROWS  (NN_*L_SEQ)   /* 4096 */
#define DBL_W 64            /* padded width of x_proj output (48 -> 64) */

// ---------------------------------------------------------------------------
// WMMA GEMM (fully tile-aligned): C[M,N] = A[M,K](lda) @ W[N,K]^T (+bias)(+add)
// Requires: M%64==0, N%64==0, K%32==0, pointers 16B-aligned rows.
// act: 0 = none, 1 = softplus. Block: 256 threads (8 wave32) -> 64x64 C tile.
// ---------------------------------------------------------------------------
__global__ __launch_bounds__(256) void gemm_wmma(
    const float* __restrict__ A, int lda,
    const float* __restrict__ W,
    const float* __restrict__ bias,
    const float* __restrict__ addsrc,
    float* __restrict__ C, int ldc,
    int M, int N, int K, int act)
{
  __shared__ _Float16 As[64][40];   // 64 rows x 32 cols (pad row to 40 halves)
  __shared__ _Float16 Bs[64][40];

  const int tid  = threadIdx.x;
  const int m0   = blockIdx.y * 64;
  const int n0   = blockIdx.x * 64;
  const int wave = tid >> 5;
  const int lane = tid & 31;
  const int wm   = wave & 3;     // M strip (16 rows each)
  const int wnh  = wave >> 2;    // N half (32 cols each)

  v8f acc0 = {}; v8f acc1 = {};

  const int lr = tid >> 2;        // staging row 0..63
  const int lc = (tid & 3) * 8;   // staging col base 0,8,16,24

  const float* Arow = A + (size_t)(m0 + lr) * lda + lc;
  const float* Wrow = W + (size_t)(n0 + lr) * K   + lc;

  for (int k0 = 0; k0 < K; k0 += 32) {
    // ---- branchless cooperative staging: 2x 16B loads -> 1x 16B LDS store --
    {
      v4f a0 = *(const v4f*)(Arow + k0);
      v4f a1 = *(const v4f*)(Arow + k0 + 4);
      v8h ah;
      #pragma unroll
      for (int j = 0; j < 4; ++j) { ah[j] = (_Float16)a0[j]; ah[4 + j] = (_Float16)a1[j]; }
      *(v8h*)&As[lr][lc] = ah;

      v4f w0 = *(const v4f*)(Wrow + k0);
      v4f w1 = *(const v4f*)(Wrow + k0 + 4);
      v8h wh;
      #pragma unroll
      for (int j = 0; j < 4; ++j) { wh[j] = (_Float16)w0[j]; wh[4 + j] = (_Float16)w1[j]; }
      *(v8h*)&Bs[lr][lc] = wh;

      if (k0 + 32 < K) {
        __builtin_prefetch(Arow + k0 + 32, 0, 1);
        __builtin_prefetch(Wrow + k0 + 32, 0, 1);
      }
    }
    __syncthreads();

    // ---- build fragments in contiguous k = 16h + s order (same for A and B) ----
    const int r   = lane & 15;
    const int akb = (lane < 16) ? 0 : 16;   // A: lanes 0-15 -> K 0-15; lanes 16-31 -> K 16-31
    v8h a_lo = *(const v8h*)&As[wm * 16 + r][akb];
    v8h a_hi = *(const v8h*)&As[wm * 16 + r][akb + 8];
    v16h a;
    #pragma unroll
    for (int j = 0; j < 8; ++j) { a[j] = a_lo[j]; a[8 + j] = a_hi[j]; }

    const int bkb = (lane < 16) ? 0 : 16;   // B: lanes 0-15 -> K 0-15; lanes 16-31 -> K 16-31
    v8h b0lo = *(const v8h*)&Bs[wnh * 32 +      r][bkb];
    v8h b0hi = *(const v8h*)&Bs[wnh * 32 +      r][bkb + 8];
    v8h b1lo = *(const v8h*)&Bs[wnh * 32 + 16 + r][bkb];
    v8h b1hi = *(const v8h*)&Bs[wnh * 32 + 16 + r][bkb + 8];
    v16h b0, b1;
    #pragma unroll
    for (int j = 0; j < 8; ++j) {
      b0[j] = b0lo[j]; b0[8 + j] = b0hi[j];
      b1[j] = b1lo[j]; b1[8 + j] = b1hi[j];
    }

    acc0 = __builtin_amdgcn_wmma_f32_16x16x32_f16(false, a, false, b0, (short)0, acc0, false, false);
    acc1 = __builtin_amdgcn_wmma_f32_16x16x32_f16(false, a, false, b1, (short)0, acc1, false, false);
    __syncthreads();
  }

  // ---- epilogue: C/D layout (VGPR j, lanes<16 -> M=j, lanes>=16 -> M=j+8) ----
  const int rbase = m0 + wm * 16 + ((lane >= 16) ? 8 : 0);
  const int c0    = n0 + wnh * 32 + (lane & 15);
  const int c1    = c0 + 16;
  const float bv0 = bias ? bias[c0] : 0.f;
  const float bv1 = bias ? bias[c1] : 0.f;
  #pragma unroll
  for (int j = 0; j < 8; ++j) {
    const int row = rbase + j;
    float v0 = acc0[j] + bv0;
    float v1 = acc1[j] + bv1;
    if (addsrc) {
      v0 += addsrc[(size_t)row * ldc + c0];
      v1 += addsrc[(size_t)row * ldc + c1];
    }
    if (act == 1) {
      v0 = (v0 > 30.f) ? v0 : log1pf(__expf(v0));
      v1 = (v1 > 30.f) ? v1 : log1pf(__expf(v1));
    }
    C[(size_t)row * ldc + c0] = v0;
    C[(size_t)row * ldc + c1] = v1;
  }
}

// ---------------------------------------------------------------------------
// Weight padding (once per launch): x_proj (NL,48,512)->(NL,64,512) zeros;
// dt_proj (NL,512,16)->(NL,512,32) zeros.
// ---------------------------------------------------------------------------
__global__ void k_pad_xproj(const float* __restrict__ src, float* __restrict__ dst)
{
  int gid = blockIdx.x * 256 + threadIdx.x;        // NL*64*512
  if (gid >= NL_ * 64 * DI_) return;
  const int k = gid & (DI_ - 1);
  const int r = (gid >> 9) & 63;
  const int l = gid >> 15;
  dst[gid] = (r < 48) ? src[((size_t)l * 48 + r) * DI_ + k] : 0.f;
}

__global__ void k_pad_dt(const float* __restrict__ src, float* __restrict__ dst)
{
  int gid = blockIdx.x * 256 + threadIdx.x;        // NL*512*32
  if (gid >= NL_ * DI_ * 32) return;
  const int k = gid & 31;
  const int r = (gid >> 5) & (DI_ - 1);
  const int l = gid >> 14;
  dst[gid] = (k < 16) ? src[((size_t)l * DI_ + r) * 16 + k] : 0.f;
}

// ---------------------------------------------------------------------------
// Embedding gather with NN-fold broadcast: x[(n*L+l)*D + d] = emb[tokens[l]*D + d]
// ---------------------------------------------------------------------------
__global__ void k_embed(const int* __restrict__ tokens, const float* __restrict__ emb,
                        float* __restrict__ x)
{
  int gid = blockIdx.x * 256 + threadIdx.x;
  if (gid >= ROWS * D_) return;
  int d   = gid & (D_ - 1);
  int row = gid >> 8;
  int l   = row & (L_SEQ - 1);
  x[gid] = emb[(size_t)tokens[l] * D_ + d];
}

// ---------------------------------------------------------------------------
// LayerNorm over D=256, one wave32 per row (8 rows per block).
// ---------------------------------------------------------------------------
__global__ __launch_bounds__(256) void k_ln(const float* __restrict__ X,
                                            const float* __restrict__ g,
                                            const float* __restrict__ b,
                                            float* __restrict__ Y, int Mrows)
{
  const int wave = threadIdx.x >> 5;
  const int lane = threadIdx.x & 31;
  const int row  = blockIdx.x * 8 + wave;
  if (row >= Mrows) return;
  const float* xr = X + (size_t)row * D_;
  float v[8]; float s = 0.f;
  #pragma unroll
  for (int j = 0; j < 8; ++j) { v[j] = xr[j * 32 + lane]; s += v[j]; }
  #pragma unroll
  for (int o = 16; o > 0; o >>= 1) s += __shfl_xor(s, o, 32);
  const float mean = s * (1.f / 256.f);
  float q = 0.f;
  #pragma unroll
  for (int j = 0; j < 8; ++j) { float dd = v[j] - mean; q += dd * dd; }
  #pragma unroll
  for (int o = 16; o > 0; o >>= 1) q += __shfl_xor(q, o, 32);
  const float rstd = rsqrtf(q * (1.f / 256.f) + 1e-5f);
  float* yr = Y + (size_t)row * D_;
  #pragma unroll
  for (int j = 0; j < 8; ++j) {
    const int c = j * 32 + lane;
    yr[c] = (v[j] - mean) * rstd * g[c] + b[c];
  }
}

// ---------------------------------------------------------------------------
// Depthwise causal conv (DC=4) + SiLU. Input = first DI cols of xz[ROWS,1024].
// ---------------------------------------------------------------------------
__global__ void k_conv_silu(const float* __restrict__ xz, const float* __restrict__ cw,
                            const float* __restrict__ cb, float* __restrict__ xm)
{
  int gid = blockIdx.x * 256 + threadIdx.x;
  if (gid >= ROWS * DI_) return;
  const int d   = gid & (DI_ - 1);
  const int row = gid >> 9;
  const int t   = row & (L_SEQ - 1);
  const int nb  = row - t;
  float acc = cb[d];
  #pragma unroll
  for (int j = 0; j < DC_; ++j) {
    const int tt = t - (DC_ - 1) + j;
    if (tt >= 0) acc += cw[d * DC_ + j] * xz[(size_t)(nb + tt) * (2 * DI_) + d];
  }
  xm[gid] = acc / (1.f + __expf(-acc));   // silu
}

// ---------------------------------------------------------------------------
// Selective scan: one thread per (n,d) channel carrying 16 states in registers.
// dbl rows have stride DBL_W; layout [dt:0-15 | B:16-31 | C:32-47 | pad].
// ---------------------------------------------------------------------------
__global__ __launch_bounds__(256) void k_scan(const float* __restrict__ dt,
                                              const float* __restrict__ dbl,
                                              const float* __restrict__ xm,
                                              const float* __restrict__ A_log,
                                              float* __restrict__ y)
{
  const int gid = blockIdx.x * 256 + threadIdx.x;   // 0..4095
  const int n = gid >> 9;
  const int d = gid & (DI_ - 1);
  float a[DS_], h[DS_];
  #pragma unroll
  for (int s = 0; s < DS_; ++s) { a[s] = -__expf(A_log[d * DS_ + s]); h[s] = 0.f; }
  const int rb = n * L_SEQ;
  for (int t = 0; t < L_SEQ; ++t) {
    const size_t row = (size_t)(rb + t);
    const float dtv = dt[row * DI_ + d];
    const float xv  = xm[row * DI_ + d];
    const float* bc = dbl + row * DBL_W;
    float acc = 0.f;
    #pragma unroll
    for (int s = 0; s < DS_; ++s) {
      const float dA = __expf(dtv * a[s]);
      h[s] = dA * h[s] + (dtv * bc[16 + s]) * xv;
      acc += h[s] * bc[32 + s];
    }
    y[row * DI_ + d] = acc;
  }
}

// ---------------------------------------------------------------------------
// Gate: y = (y + xm*Dp) * silu(z), z = second half of xz row.
// ---------------------------------------------------------------------------
__global__ void k_gate(float* __restrict__ y, const float* __restrict__ xm,
                       const float* __restrict__ xz, const float* __restrict__ Dp)
{
  int gid = blockIdx.x * 256 + threadIdx.x;
  if (gid >= ROWS * DI_) return;
  const int d   = gid & (DI_ - 1);
  const int row = gid >> 9;
  const float z  = xz[(size_t)row * (2 * DI_) + DI_ + d];
  const float sz = z / (1.f + __expf(-z));
  y[gid] = (y[gid] + xm[gid] * Dp[d]) * sz;
}

// ---------------------------------------------------------------------------
// Tiny attention across NN=8 replicas. One thread per (l, h, i) query row.
// qkv rows are (n*L + l), cols [q:0..255 | k:256..511 | v:512..767].
// ---------------------------------------------------------------------------
__global__ void k_attn(const float* __restrict__ qkv, float* __restrict__ atto)
{
  int gid = blockIdx.x * 256 + threadIdx.x;
  if (gid >= L_SEQ * H_ * NN_) return;
  const int i = gid & 7;
  const int h = (gid >> 3) & 3;
  const int l = gid >> 5;
  const int dh = 64;
  const float* q = qkv + (size_t)(i * L_SEQ + l) * (3 * D_) + h * dh;
  float sc[NN_];
  float mx = -1e30f;
  for (int j = 0; j < NN_; ++j) {
    const float* k = qkv + (size_t)(j * L_SEQ + l) * (3 * D_) + D_ + h * dh;
    float s = 0.f;
    #pragma unroll 8
    for (int d = 0; d < dh; ++d) s += q[d] * k[d];
    s *= 0.125f;                        // 1/sqrt(64)
    sc[j] = s; mx = fmaxf(mx, s);
  }
  float denom = 0.f;
  #pragma unroll
  for (int j = 0; j < NN_; ++j) { sc[j] = __expf(sc[j] - mx); denom += sc[j]; }
  const float inv = 1.f / denom;
  float* o = atto + (size_t)(i * L_SEQ + l) * D_ + h * dh;
  for (int d = 0; d < dh; ++d) {
    float acc = 0.f;
    #pragma unroll
    for (int j = 0; j < NN_; ++j)
      acc += sc[j] * qkv[(size_t)(j * L_SEQ + l) * (3 * D_) + 2 * D_ + h * dh + d];
    o[d] = acc * inv;
  }
}

// ---------------------------------------------------------------------------
// Mean over NN replicas: xf[l,d] = mean_n x[(n*L+l), d]
// ---------------------------------------------------------------------------
__global__ void k_mean(const float* __restrict__ x, float* __restrict__ xf)
{
  int gid = blockIdx.x * 256 + threadIdx.x;
  if (gid >= L_SEQ * D_) return;
  const int d = gid & (D_ - 1);
  const int l = gid >> 8;
  float s = 0.f;
  #pragma unroll
  for (int n = 0; n < NN_; ++n) s += x[(size_t)(n * L_SEQ + l) * D_ + d];
  xf[gid] = s * (1.f / NN_);
}

// ---------------------------------------------------------------------------
extern "C" void kernel_launch(void* const* d_in, const int* in_sizes, int n_in,
                              void* d_out, int out_size, void* d_ws, size_t ws_size,
                              hipStream_t stream)
{
  const int*   tokens     = (const int*)  d_in[0];
  const float* emb        = (const float*)d_in[1];
  const float* head_b     = (const float*)d_in[2];
  const float* in_proj_w  = (const float*)d_in[3];
  const float* conv_w     = (const float*)d_in[4];
  const float* conv_b     = (const float*)d_in[5];
  const float* x_proj_w   = (const float*)d_in[6];
  const float* dt_proj_w  = (const float*)d_in[7];
  const float* dt_proj_b  = (const float*)d_in[8];
  const float* A_log      = (const float*)d_in[9];
  const float* D_param    = (const float*)d_in[10];
  const float* out_proj_w = (const float*)d_in[11];
  const float* attn_in_w  = (const float*)d_in[12];
  const float* attn_in_b  = (const float*)d_in[13];
  const float* attn_out_w = (const float*)d_in[14];
  const float* attn_out_b = (const float*)d_in[15];
  const float* n1_g       = (const float*)d_in[16];
  const float* n1_b       = (const float*)d_in[17];
  const float* n2_g       = (const float*)d_in[18];
  const float* n2_b       = (const float*)d_in[19];
  const float* normf_g    = (const float*)d_in[20];
  const float* normf_b    = (const float*)d_in[21];

  float* ws  = (float*)d_ws;
  float* x    = ws;                                   // ROWS*D
  float* xn   = x    + (size_t)ROWS * D_;             // ROWS*D
  float* xz   = xn   + (size_t)ROWS * D_;             // ROWS*1024
  float* xm   = xz   + (size_t)ROWS * 2 * DI_;        // ROWS*DI
  float* dbl  = xm   + (size_t)ROWS * DI_;            // ROWS*DBL_W
  float* dt   = dbl  + (size_t)ROWS * DBL_W;          // ROWS*DI
  float* y    = dt   + (size_t)ROWS * DI_;            // ROWS*DI
  float* wxp  = y    + (size_t)ROWS * DI_;            // NL*64*DI   (padded x_proj_w)
  float* wdt  = wxp  + (size_t)NL_ * 64 * DI_;        // NL*DI*32   (padded dt_proj_w)
  float* qkv  = xz;    // alias: xz dead once gate has consumed z
  float* atto = dt;    // alias: dt dead after scan
  float* xf   = dbl;   // alias: only used at the end

  const dim3 blk(256);

  k_pad_xproj<<<(NL_ * 64 * DI_ + 255) / 256, blk, 0, stream>>>(x_proj_w, wxp);
  k_pad_dt   <<<(NL_ * DI_ * 32 + 255) / 256, blk, 0, stream>>>(dt_proj_w, wdt);
  k_embed<<<(ROWS * D_ + 255) / 256, blk, 0, stream>>>(tokens, emb, x);

  for (int i = 0; i < NL_; ++i) {
    const float* W_in  = in_proj_w  + (size_t)i * 2 * DI_ * D_;
    const float* cw    = conv_w     + (size_t)i * DI_ * DC_;
    const float* cb    = conv_b     + (size_t)i * DI_;
    const float* W_xp  = wxp        + (size_t)i * 64 * DI_;     // padded, N=64
    const float* W_dt  = wdt        + (size_t)i * DI_ * 32;     // padded, K=32
    const float* b_dt  = dt_proj_b  + (size_t)i * DI_;
    const float* Al    = A_log      + (size_t)i * DI_ * DS_;
    const float* Dp    = D_param    + (size_t)i * DI_;
    const float* W_out = out_proj_w + (size_t)i * D_ * DI_;
    const float* W_ai  = attn_in_w  + (size_t)i * 3 * D_ * D_;
    const float* b_ai  = attn_in_b  + (size_t)i * 3 * D_;
    const float* W_ao  = attn_out_w + (size_t)i * D_ * D_;
    const float* b_ao  = attn_out_b + (size_t)i * D_;

    // ---- Mamba branch ----
    k_ln<<<ROWS / 8, blk, 0, stream>>>(x, n1_g + i * D_, n1_b + i * D_, xn, ROWS);
    gemm_wmma<<<dim3((2 * DI_) / 64, ROWS / 64), blk, 0, stream>>>(
        xn, D_, W_in, nullptr, nullptr, xz, 2 * DI_, ROWS, 2 * DI_, D_, 0);
    k_conv_silu<<<(ROWS * DI_ + 255) / 256, blk, 0, stream>>>(xz, cw, cb, xm);
    gemm_wmma<<<dim3(1, ROWS / 64), blk, 0, stream>>>(
        xm, DI_, W_xp, nullptr, nullptr, dbl, DBL_W, ROWS, 64, DI_, 0);
    gemm_wmma<<<dim3(DI_ / 64, ROWS / 64), blk, 0, stream>>>(
        dbl, DBL_W, W_dt, b_dt, nullptr, dt, DI_, ROWS, DI_, 32, /*softplus*/1);
    k_scan<<<ROWS / 256, blk, 0, stream>>>(dt, dbl, xm, Al, y);
    k_gate<<<(ROWS * DI_ + 255) / 256, blk, 0, stream>>>(y, xm, xz, Dp);
    gemm_wmma<<<dim3(D_ / 64, ROWS / 64), blk, 0, stream>>>(
        y, DI_, W_out, nullptr, /*residual*/x, x, D_, ROWS, D_, DI_, 0);

    // ---- Attention branch (across NN replicas) ----
    k_ln<<<ROWS / 8, blk, 0, stream>>>(x, n2_g + i * D_, n2_b + i * D_, xn, ROWS);
    gemm_wmma<<<dim3((3 * D_) / 64, ROWS / 64), blk, 0, stream>>>(
        xn, D_, W_ai, b_ai, nullptr, qkv, 3 * D_, ROWS, 3 * D_, D_, 0);
    k_attn<<<(L_SEQ * H_ * NN_ + 255) / 256, blk, 0, stream>>>(qkv, atto);
    gemm_wmma<<<dim3(D_ / 64, ROWS / 64), blk, 0, stream>>>(
        atto, D_, W_ao, b_ao, /*residual*/x, x, D_, ROWS, D_, D_, 0);
  }

  // ---- Head ----
  k_mean<<<(L_SEQ * D_ + 255) / 256, blk, 0, stream>>>(x, xf);
  k_ln<<<L_SEQ / 8, blk, 0, stream>>>(xf, normf_g, normf_b, xf, L_SEQ);
  gemm_wmma<<<dim3(32000 / 64, L_SEQ / 64), blk, 0, stream>>>(
      xf, D_, emb, head_b, nullptr, (float*)d_out, 32000, L_SEQ, 32000, D_, 0);
}
